// Model_8443905704627
// MI455X (gfx1250) — compile-verified, matrix operand order checked by reading, not run on hardware
//
#include <hip/hip_runtime.h>

typedef __attribute__((ext_vector_type(16))) __bf16 v16bf;
typedef __attribute__((ext_vector_type(8)))  float  v8f;
typedef __attribute__((ext_vector_type(4)))  unsigned int v4u;

#define LSEQ 4096
#define DDIM 128
#define NBATCH 4

static constexpr float kScale = 0.08838834764831845f; // 1/sqrt(128)

union FragU { v4u u[2]; v16bf v; };

// Load a 16x32 bf16 fragment (A-layout / B-as-transposed-A layout) from a
// row-major tile with row stride `ld` elements. One lane: row = lane&15,
// k-base = (lane>>4)*8, halves at +0 and +16. Two 16-byte loads per lane.
__device__ __forceinline__ v16bf load_frag(const __bf16* tile, int ld, int lane) {
    const int m  = lane & 15;
    const int kb = (lane >> 4) * 8;
    FragU f;
    f.u[0] = *(const v4u*)(tile + (size_t)m * ld + kb);
    f.u[1] = *(const v4u*)(tile + (size_t)m * ld + kb + 16);
    return f.v;
}

// Wave-relative LDS byte offset of a shared-memory object (what the async
// global->LDS instructions expect in their LDS-address VGPR).
__device__ __forceinline__ unsigned lds_off(const void* p) {
    return (unsigned)(uintptr_t)(__attribute__((address_space(3))) const void*)p;
}

// Y = X * W^T + b  (torch Linear). X: [M,128] f32, W: [128,128] f32 row-major.
// Output bf16, either [M,128] (transpose=0) or per-batch transposed
// [B][128][LSEQ] (transpose=1, used for V so PV WMMA can load B-frags).
__global__ __launch_bounds__(256) void proj_kernel(
    const float* __restrict__ X, const float* __restrict__ W,
    const float* __restrict__ bias, __bf16* __restrict__ Y, int transpose) {
    __shared__ __bf16 Wlds[DDIM * DDIM];   // 32 KB, bf16 copy of W
    __shared__ float  blds[DDIM];

    const int tid  = threadIdx.x;
    const int lane = tid & 31;
    const int wid  = tid >> 5;

    // Cooperative convert W (f32 -> bf16) into LDS.
    for (int i = tid * 4; i < DDIM * DDIM; i += 256 * 4) {
        float4 w4 = *(const float4*)(W + i);
        Wlds[i + 0] = (__bf16)w4.x;
        Wlds[i + 1] = (__bf16)w4.y;
        Wlds[i + 2] = (__bf16)w4.z;
        Wlds[i + 3] = (__bf16)w4.w;
    }
    if (tid < DDIM) blds[tid] = bias[tid];
    __syncthreads();

    const int row0  = blockIdx.x * 128 + wid * 16;
    const int m     = lane & 15;
    const int kb8   = (lane >> 4) * 8;
    const int h     = lane >> 4;
    const int nlane = lane & 15;

    // Load this wave's 16x128 X strip as 4 bf16 A-fragments.
    v16bf afrag[4];
    const float* xrow = X + (size_t)(row0 + m) * DDIM;
#pragma unroll
    for (int c = 0; c < 4; ++c) {
        const int base = c * 32 + kb8;
        float4 x0 = *(const float4*)(xrow + base);
        float4 x1 = *(const float4*)(xrow + base + 4);
        float4 x2 = *(const float4*)(xrow + base + 16);
        float4 x3 = *(const float4*)(xrow + base + 20);
        v16bf a;
        a[0] = (__bf16)x0.x; a[1] = (__bf16)x0.y; a[2] = (__bf16)x0.z; a[3] = (__bf16)x0.w;
        a[4] = (__bf16)x1.x; a[5] = (__bf16)x1.y; a[6] = (__bf16)x1.z; a[7] = (__bf16)x1.w;
        a[8]  = (__bf16)x2.x; a[9]  = (__bf16)x2.y; a[10] = (__bf16)x2.z; a[11] = (__bf16)x2.w;
        a[12] = (__bf16)x3.x; a[13] = (__bf16)x3.y; a[14] = (__bf16)x3.z; a[15] = (__bf16)x3.w;
        afrag[c] = a;
    }

#pragma unroll
    for (int nt = 0; nt < 8; ++nt) {
        const float bn = blds[nt * 16 + nlane];
        v8f acc;
#pragma unroll
        for (int i = 0; i < 8; ++i) acc[i] = bn;
#pragma unroll
        for (int c = 0; c < 4; ++c) {
            v16bf bfrag = load_frag(&Wlds[(size_t)(nt * 16) * DDIM + c * 32], DDIM, lane);
            acc = __builtin_amdgcn_wmma_f32_16x16x32_bf16(
                false, afrag[c], false, bfrag, (short)0, acc, false, false);
        }
        if (!transpose) {
            __bf16* yp = Y + (size_t)(row0 + 8 * h) * DDIM + nt * 16 + nlane;
#pragma unroll
            for (int r = 0; r < 8; ++r) yp[(size_t)r * DDIM] = (__bf16)acc[r];
        } else {
            const int Rbase = row0 + 8 * h;           // global row -> (batch, t)
            const int b = Rbase / LSEQ;
            const int t = Rbase % LSEQ;
            __bf16* yp = Y + ((size_t)b * DDIM + nt * 16 + nlane) * LSEQ + t;
#pragma unroll
            for (int r = 0; r < 8; ++r) yp[r] = (__bf16)acc[r];
        }
    }
}

// Async-DMA stage one 32-column chunk: K tile 32x128 (8 KB contiguous) and
// V^T tile 128x32 (128 rows x 64 B) into LDS. 4 async b128 loads per thread,
// tracked on ASYNCcnt (no VGPR data, no dscnt/loadcnt).
__device__ __forceinline__ void stage_chunk(
    unsigned klds, unsigned vlds,
    const __bf16* kchunk, const __bf16* vchunk, int tid) {
    const unsigned kg0 = (unsigned)tid * 16u;
    const unsigned kg1 = kg0 + 4096u;
    asm volatile("global_load_async_to_lds_b128 %0, %1, %2"
                 :: "v"(klds + kg0), "v"(kg0), "s"(kchunk) : "memory");
    asm volatile("global_load_async_to_lds_b128 %0, %1, %2"
                 :: "v"(klds + kg1), "v"(kg1), "s"(kchunk) : "memory");
#pragma unroll
    for (int it = 0; it < 2; ++it) {
        const int i = tid + it * 256;                 // i -> (n = i>>2, q = i&3)
        const unsigned n = (unsigned)(i >> 2), q = (unsigned)(i & 3);
        const unsigned goff = (n * (unsigned)LSEQ + q * 8u) * 2u;
        const unsigned loff = (unsigned)i * 16u;
        asm volatile("global_load_async_to_lds_b128 %0, %1, %2"
                     :: "v"(vlds + loff), "v"(goff), "s"(vchunk) : "memory");
    }
}

// Flash attention: Q [B][L][128] bf16, K [B][L][128] bf16, VT [B][128][L] bf16.
// Block: 8 waves x 16 Q rows = 128 rows. Double-buffered async K/V staging
// overlapped with WMMA compute; one barrier per 32-column chunk.
__global__ __launch_bounds__(256) void flash_kernel(
    const __bf16* __restrict__ Q, const __bf16* __restrict__ K,
    const __bf16* __restrict__ VT, float* __restrict__ O) {
    __shared__ __bf16 Klds[2][32 * DDIM];   // 2 x 8 KB : K chunk, rows t, cols d
    __shared__ __bf16 Vlds[2][DDIM * 32];   // 2 x 8 KB : V^T chunk, rows n, cols t
    __shared__ __bf16 Plds[8][16 * 32];     // 8 KB     : per-wave P relayout buffer

    const int tid   = threadIdx.x;
    const int lane  = tid & 31;
    const int wid   = tid >> 5;
    const int b     = blockIdx.y;
    const int qrow  = blockIdx.x * 128 + wid * 16;
    const int m     = lane & 15;
    const int kb8   = (lane >> 4) * 8;
    const int h     = lane >> 4;
    const int nlane = lane & 15;

    // Q fragments held in registers for the whole pass (4 x 16x32 bf16).
    v16bf qfrag[4];
    const __bf16* qrowp = Q + ((size_t)b * LSEQ + qrow + m) * DDIM;
#pragma unroll
    for (int c = 0; c < 4; ++c) {
        FragU f;
        f.u[0] = *(const v4u*)(qrowp + c * 32 + kb8);
        f.u[1] = *(const v4u*)(qrowp + c * 32 + kb8 + 16);
        qfrag[c] = f.v;
    }

    v8f oacc[8];
#pragma unroll
    for (int nt = 0; nt < 8; ++nt)
#pragma unroll
        for (int i = 0; i < 8; ++i) oacc[nt][i] = 0.0f;
    float rowmax[8], rowsum[8];
#pragma unroll
    for (int r = 0; r < 8; ++r) { rowmax[r] = -__builtin_inff(); rowsum[r] = 0.0f; }

    const __bf16* kbase = K  + (size_t)b * LSEQ * DDIM;
    const __bf16* vbase = VT + (size_t)b * DDIM * LSEQ;

    const unsigned klds0 = lds_off(&Klds[0][0]), klds1 = lds_off(&Klds[1][0]);
    const unsigned vlds0 = lds_off(&Vlds[0][0]), vlds1 = lds_off(&Vlds[1][0]);

    // Prologue: stage chunk 0 into buffer 0.
    stage_chunk(klds0, vlds0, kbase, vbase, tid);

    constexpr int NCHUNK = LSEQ / 32;
    for (int it = 0; it < NCHUNK; ++it) {
        const int buf = it & 1;
        // Own async writes for `buf` complete; barrier then makes all waves'
        // writes visible and guarantees nobody still reads the other buffer.
        asm volatile("s_wait_asynccnt 0x0" ::: "memory");
        __syncthreads();
        if (it + 1 < NCHUNK) {
            const int t1 = (it + 1) * 32;
            stage_chunk(buf ? klds0 : klds1, buf ? vlds0 : vlds1,
                        kbase + (size_t)t1 * DDIM, vbase + t1, tid);
        }
        const __bf16* Kb = &Klds[buf][0];
        const __bf16* Vb = &Vlds[buf][0];

        // ---- S = Q K^T for 2 column tiles (16 t each), K reduced over 128 ----
        v8f s0, s1;
#pragma unroll
        for (int i = 0; i < 8; ++i) { s0[i] = 0.0f; s1[i] = 0.0f; }
#pragma unroll
        for (int c = 0; c < 4; ++c) {
            v16bf k0 = load_frag(Kb + c * 32, DDIM, lane);
            v16bf k1 = load_frag(Kb + 16 * DDIM + c * 32, DDIM, lane);
            s0 = __builtin_amdgcn_wmma_f32_16x16x32_bf16(
                false, qfrag[c], false, k0, (short)0, s0, false, false);
            s1 = __builtin_amdgcn_wmma_f32_16x16x32_bf16(
                false, qfrag[c], false, k1, (short)0, s1, false, false);
        }
#pragma unroll
        for (int i = 0; i < 8; ++i) { s0[i] *= kScale; s1[i] *= kScale; }

        // ---- Online softmax over this 32-column chunk ----
        // C layout: VGPR r + lane-half h -> row r+8h; lanes within a half hold
        // the 16 columns, so row reductions stay inside 16-lane groups.
        __bf16* pl = &Plds[wid][0];
#pragma unroll
        for (int r = 0; r < 8; ++r) {
            float tmax = fmaxf(s0[r], s1[r]);
            tmax = fmaxf(tmax, __shfl_xor(tmax, 1, 32));
            tmax = fmaxf(tmax, __shfl_xor(tmax, 2, 32));
            tmax = fmaxf(tmax, __shfl_xor(tmax, 4, 32));
            tmax = fmaxf(tmax, __shfl_xor(tmax, 8, 32));
            const float mnew = fmaxf(rowmax[r], tmax);
            const float corr = __expf(rowmax[r] - mnew);
            rowmax[r] = mnew;
            const float p0 = __expf(s0[r] - mnew);
            const float p1 = __expf(s1[r] - mnew);
            float ssum = p0 + p1;
            ssum += __shfl_xor(ssum, 1, 32);
            ssum += __shfl_xor(ssum, 2, 32);
            ssum += __shfl_xor(ssum, 4, 32);
            ssum += __shfl_xor(ssum, 8, 32);
            rowsum[r] = rowsum[r] * corr + ssum;
#pragma unroll
            for (int nt = 0; nt < 8; ++nt) oacc[nt][r] *= corr;
            const int prow = r + 8 * h;
            pl[prow * 32 + nlane]      = (__bf16)p0;
            pl[prow * 32 + 16 + nlane] = (__bf16)p1;
        }
        asm volatile("s_wait_dscnt 0" ::: "memory");  // P stores -> P frag load (same wave)
        v16bf pfrag = load_frag(pl, 32, lane);

        // ---- O += P V : 8 output n-tiles, B-frags from V^T rows ----
#pragma unroll
        for (int nt = 0; nt < 8; ++nt) {
            v16bf vfrag = load_frag(Vb + (size_t)(nt * 16) * 32, 32, lane);
            oacc[nt] = __builtin_amdgcn_wmma_f32_16x16x32_bf16(
                false, pfrag, false, vfrag, (short)0, oacc[nt], false, false);
        }
        // No trailing barrier: next iteration's wait+barrier precedes any
        // overwrite of this buffer.
    }

    // ---- Normalize and write f32 output ----
    float rinv[8];
#pragma unroll
    for (int r = 0; r < 8; ++r) rinv[r] = 1.0f / rowsum[r];
    float* orow = O + ((size_t)b * LSEQ + qrow + 8 * h) * DDIM + nlane;
#pragma unroll
    for (int nt = 0; nt < 8; ++nt)
#pragma unroll
        for (int r = 0; r < 8; ++r)
            orow[(size_t)r * DDIM + nt * 16] = oacc[nt][r] * rinv[r];
}

extern "C" void kernel_launch(void* const* d_in, const int* in_sizes, int n_in,
                              void* d_out, int out_size, void* d_ws, size_t ws_size,
                              hipStream_t stream) {
    (void)in_sizes; (void)n_in; (void)out_size; (void)ws_size;
    const float* query = (const float*)d_in[0];
    const float* key   = (const float*)d_in[1];
    const float* value = (const float*)d_in[2];
    const float* Wq    = (const float*)d_in[3];
    const float* bq    = (const float*)d_in[4];
    const float* Wk    = (const float*)d_in[5];
    const float* bk    = (const float*)d_in[6];
    const float* Wv    = (const float*)d_in[7];
    const float* bv    = (const float*)d_in[8];
    float* out = (float*)d_out;

    const size_t M = (size_t)NBATCH * LSEQ;           // 16384 rows
    __bf16* qb  = (__bf16*)d_ws;                      // [B][L][128]
    __bf16* kb  = qb + M * DDIM;                      // [B][L][128]
    __bf16* vtb = kb + M * DDIM;                      // [B][128][L]

    const int pblocks = (int)(M / 128);               // 128
    proj_kernel<<<pblocks, 256, 0, stream>>>(query, Wq, bq, qb, 0);
    proj_kernel<<<pblocks, 256, 0, stream>>>(key,   Wk, bk, kb, 0);
    proj_kernel<<<pblocks, 256, 0, stream>>>(value, Wv, bv, vtb, 1);

    dim3 grid(LSEQ / 128, NBATCH);                    // (32, 4)
    flash_kernel<<<grid, 256, 0, stream>>>(qb, kb, vtb, out);
}
